// GoldFinchTimeMix_7060926234674
// MI455X (gfx1250) — matrix operand order repaired and checked by reading, not compiled, not run on hardware
//
#include <hip/hip_runtime.h>
#include <hip/hip_bf16.h>

// GoldFinch TimeMix for gfx1250 (MI455X).
//  1) transpose W_r/W_k/W_v/W_o -> bf16 N x K (WMMA B-fragment friendly)
//  2) prep_mix: token-shift, 160-wide mix MLP, 5 mixed activations (bf16),
//     decay = exp(-exp(w)) and v2 correction (64-wide MLPs) in f32
//  3) 4x WMMA bf16 GEMM (8192x2048x2048), 64x64 per-wave tiles, fused epilogues
//  4) WKV scan: 64x64 state in VGPRs, one block per (b,h), software-pipelined loads
//  5) layernorm -> bf16, then WMMA GEMM with W_o into d_out
//  6) shift_state copy

typedef __bf16 bf16;
typedef __attribute__((ext_vector_type(16))) bf16  bf16x16;
typedef __attribute__((ext_vector_type(8)))  bf16  bf16x8;
typedef __attribute__((ext_vector_type(8)))  float f32x8;

#define DD   2048
#define BB   4
#define TT   2048
#define HH   32
#define KHD  64
#define NMIXC 5
#define DMIXC 32
#define DDEC  64

// ---------------------------------------------------------------- transpose
// src: K x N f32 row-major  ->  dst: N x K bf16 row-major
__global__ __launch_bounds__(256) void gf_transpose_bf16(
    const float* __restrict__ src, bf16* __restrict__ dst, int K, int N) {
  __shared__ float tile[16][17];
  int n = blockIdx.x * 16 + threadIdx.x;
  int k = blockIdx.y * 16 + threadIdx.y;
  tile[threadIdx.y][threadIdx.x] = src[(size_t)k * N + n];
  __syncthreads();
  int n2 = blockIdx.x * 16 + threadIdx.y;
  int k2 = blockIdx.y * 16 + threadIdx.x;
  dst[(size_t)n2 * K + k2] = (bf16)tile[threadIdx.x][threadIdx.y];
}

// ---------------------------------------------------------------- prep_mix
__global__ __launch_bounds__(256) void gf_prep_mix(
    const float* __restrict__ x, const float* __restrict__ shift,
    const float* __restrict__ tmx, const float* __restrict__ tmr,
    const float* __restrict__ tmk, const float* __restrict__ tmv,
    const float* __restrict__ tmw, const float* __restrict__ tmv2,
    const float* __restrict__ w1,   // D x 160
    const float* __restrict__ w2,   // 160 x D (flattened 5x32xD)
    const float* __restrict__ tdec, // D
    const float* __restrict__ tdw1, // D x 64
    const float* __restrict__ tdw2, // 64 x D
    const float* __restrict__ tvw1, // D x 64
    const float* __restrict__ tvw2, // 64 x D
    bf16* __restrict__ xrO, bf16* __restrict__ xkO, bf16* __restrict__ xvO,
    bf16* __restrict__ xv2O, float* __restrict__ decayO, float* __restrict__ v2cO) {
  __shared__ float lx[DD], ldx[DD], la[DD], lv2[DD];
  __shared__ float t160[NMIXC * DMIXC];
  __shared__ float t64w[DDEC], t64v[DDEC];

  const int token = blockIdx.x;
  const int b = token / TT, t = token % TT;
  const int tid = threadIdx.x;
  const float* xrow = x + (size_t)token * DD;
  const float* xp = (t == 0) ? (shift + (size_t)b * DD)
                             : (x + ((size_t)token - 1) * DD);

  // stage 1: token shift + xxx
  for (int d = tid; d < DD; d += 256) {
    float xv_ = xrow[d];
    float dx = xp[d] - xv_;
    lx[d] = xv_; ldx[d] = dx;
    la[d] = xv_ + dx * tmx[d];
  }
  __syncthreads();

  // stage 2: t160 = tanh(xxx @ w1)
  if (tid < NMIXC * DMIXC) {
    float acc = 0.f;
    for (int i = 0; i < DD; ++i) acc += la[i] * w1[(size_t)i * (NMIXC * DMIXC) + tid];
    t160[tid] = tanhf(acc);
  }
  __syncthreads();

  // stage 3: five mixed activations
  for (int f = 0; f < NMIXC; ++f) {
    const float* maa = (f == 0) ? tmr : (f == 1) ? tmk : (f == 2) ? tmv
                       : (f == 3) ? tmw : tmv2;
    for (int d = tid; d < DD; d += 256) {
      float m = 0.f;
      #pragma unroll
      for (int j = 0; j < DMIXC; ++j)
        m += t160[f * DMIXC + j] * w2[(size_t)(f * DMIXC + j) * DD + d];
      float val = lx[d] + ldx[d] * (maa[d] + m);
      if (f == 0)      xrO[(size_t)token * DD + d] = (bf16)val;
      else if (f == 1) xkO[(size_t)token * DD + d] = (bf16)val;
      else if (f == 2) xvO[(size_t)token * DD + d] = (bf16)val;
      else if (f == 3) la[d] = val;                 // xw (reuse xxx buffer)
      else { lv2[d] = val; xv2O[(size_t)token * DD + d] = (bf16)val; }
    }
  }
  __syncthreads();

  // stage 4: small MLP hidden layers (threads 0..63 -> decay, 64..127 -> v2)
  if (tid < DDEC) {
    float acc = 0.f;
    for (int i = 0; i < DD; ++i) acc += la[i] * tdw1[(size_t)i * DDEC + tid];
    t64w[tid] = tanhf(acc);
  } else if (tid < 2 * DDEC) {
    int j = tid - DDEC;
    float acc = 0.f;
    for (int i = 0; i < DD; ++i) acc += lv2[i] * tvw1[(size_t)i * DDEC + j];
    t64v[j] = tanhf(acc);
  }
  __syncthreads();

  // stage 5: decay and v2 correction
  for (int d = tid; d < DD; d += 256) {
    float w = tdec[d], vc = 0.f;
    #pragma unroll
    for (int j = 0; j < DDEC; ++j) {
      w  += t64w[j] * tdw2[(size_t)j * DD + d];
      vc += t64v[j] * tvw2[(size_t)j * DD + d];
    }
    decayO[(size_t)token * DD + d] = expf(-expf(w));
    v2cO[(size_t)token * DD + d] = vc;
  }
}

// ---------------------------------------------------------------- WMMA GEMM
// C[M,N] = A[M,K] @ B (B given transposed: Bt[N,K]), bf16 in, f32 out.
// 8 waves/block arranged 2(M) x 4(N); each wave computes a 64x64 tile
// (4x4 WMMA 16x16x32 accumulators). Block tile = 128 x 256.
// mode 0: plain   mode 1: C *= (1 - aux)   mode 2: C += aux
__global__ __launch_bounds__(256) void gf_gemm_bf16(
    const bf16* __restrict__ A, const bf16* __restrict__ Bt,
    float* __restrict__ C, int M, int N, int K, int mode,
    const float* __restrict__ aux) {
  const int wave = threadIdx.x >> 5;
  const int lane = threadIdx.x & 31;
  const int wm = wave & 1, wn = wave >> 1;              // 2 x 4 waves
  const int mblk = blockIdx.x * 128 + wm * 64;          // 64 rows / wave
  const int nblk = blockIdx.y * 256 + wn * 64;          // 64 cols / wave
  const int lr = lane & 15, lh = lane >> 4;

  f32x8 acc[4][4] = {};

  const bf16* arow[4];
  const bf16* brow[4];
  #pragma unroll
  for (int i = 0; i < 4; ++i) arow[i] = A + (size_t)(mblk + i * 16 + lr) * K;
  #pragma unroll
  for (int j = 0; j < 4; ++j) brow[j] = Bt + (size_t)(nblk + j * 16 + lr) * K;

  union V16 { bf16x16 v; bf16x8 h[2]; };

  for (int kk = 0; kk < K; kk += 32) {
    V16 a[4];
    bf16x16 bv[4];
    #pragma unroll
    for (int i = 0; i < 4; ++i) {
      a[i].v = *(const bf16x16*)(arow[i] + kk + lh * 16);
    }
    #pragma unroll
    for (int j = 0; j < 4; ++j)
      bv[j] = *(const bf16x16*)(brow[j] + kk + lh * 16);

    #pragma unroll
    for (int i = 0; i < 4; ++i)
      #pragma unroll
      for (int j = 0; j < 4; ++j)
        acc[i][j] = __builtin_amdgcn_wmma_f32_16x16x32_bf16(
            false, a[i].v, false, bv[j], (short)0, acc[i][j], false, false);
  }

  #pragma unroll
  for (int i = 0; i < 4; ++i)
    #pragma unroll
    for (int j = 0; j < 4; ++j)
      #pragma unroll
      for (int g = 0; g < 8; ++g) {
        int row = mblk + i * 16 + lh * 8 + g;
        int col = nblk + j * 16 + lr;
        size_t idx = (size_t)row * N + col;
        float v = acc[i][j][g];
        if (mode == 1)      v *= (1.0f - aux[idx]);
        else if (mode == 2) v += aux[idx];
        C[idx] = v;
      }
}

// ---------------------------------------------------------------- WKV scan
// One block (64 threads) per (b,h); 64x64 state held in VGPRs (column per
// thread). Loads for step t+1 are issued before the step-t update so global
// latency overlaps the sequential-compute critical path.
__global__ __launch_bounds__(64) void gf_wkv_scan(
    const float* __restrict__ r, const float* __restrict__ k,
    const float* __restrict__ v, const float* __restrict__ decay,
    const float* __restrict__ v2, const float* __restrict__ Sin,
    float* __restrict__ y, float* __restrict__ Sout) {
  const int bh = blockIdx.x;            // b*H + h
  const int b = bh / HH, h = bh % HH;
  const int vc = threadIdx.x;           // state column 0..63
  float s[KHD];
  #pragma unroll
  for (int kk = 0; kk < KHD; ++kk)
    s[kk] = Sin[((size_t)bh * KHD + kk) * KHD + vc];

  const size_t base = (size_t)b * TT * DD + h * KHD + vc;
  float rn = r[base], kn = k[base], dn = decay[base], vn = v[base], v2n = v2[base];

  __shared__ float rs[KHD], ks[KHD], ds[KHD];
  for (int t = 0; t < TT; ++t) {
    rs[vc] = rn; ks[vc] = kn; ds[vc] = dn;
    float vv = vn, v2cur = v2n;
    if (t + 1 < TT) {                    // prefetch next step
      size_t nb = base + (size_t)(t + 1) * DD;
      rn = r[nb]; kn = k[nb]; dn = decay[nb]; vn = v[nb]; v2n = v2[nb];
    }
    __syncthreads();
    float acc = 0.f;
    #pragma unroll
    for (int kk = 0; kk < KHD; ++kk) {
      acc += rs[kk] * s[kk];            // y from pre-update state
      s[kk] = s[kk] * ds[kk] + ks[kk] * vv;
    }
    y[base + (size_t)t * DD] = acc + v2cur;
    __syncthreads();
  }
  #pragma unroll
  for (int kk = 0; kk < KHD; ++kk)
    Sout[((size_t)bh * KHD + kk) * KHD + vc] = s[kk];
}

// ---------------------------------------------------------------- layernorm
__global__ __launch_bounds__(256) void gf_layernorm(
    const float* __restrict__ y, const float* __restrict__ w,
    const float* __restrict__ bb, bf16* __restrict__ out) {
  __shared__ float s1[256], s2[256];
  const int token = blockIdx.x, tid = threadIdx.x;
  const float* row = y + (size_t)token * DD;
  float a = 0.f, q = 0.f;
  for (int d = tid; d < DD; d += 256) { float v0 = row[d]; a += v0; q += v0 * v0; }
  s1[tid] = a; s2[tid] = q; __syncthreads();
  for (int off = 128; off > 0; off >>= 1) {
    if (tid < off) { s1[tid] += s1[tid + off]; s2[tid] += s2[tid + off]; }
    __syncthreads();
  }
  float mu = s1[0] * (1.0f / DD);
  float var = s2[0] * (1.0f / DD) - mu * mu;
  float inv = rsqrtf(var + 1e-5f);
  for (int d = tid; d < DD; d += 256)
    out[(size_t)token * DD + d] = (bf16)((row[d] - mu) * inv * w[d] + bb[d]);
}

// ---------------------------------------------------------------- shift copy
__global__ __launch_bounds__(256) void gf_shift_copy(
    const float* __restrict__ x, float* __restrict__ so) {
  int i = blockIdx.x * 256 + threadIdx.x;
  if (i < BB * DD) {
    int b = i / DD, d = i % DD;
    so[i] = x[((size_t)b * TT + (TT - 1)) * DD + d];
  }
}

// ---------------------------------------------------------------- launcher
extern "C" void kernel_launch(void* const* d_in, const int* in_sizes, int n_in,
                              void* d_out, int out_size, void* d_ws, size_t ws_size,
                              hipStream_t stream) {
  (void)in_sizes; (void)n_in; (void)out_size; (void)ws_size;
  const float* x      = (const float*)d_in[0];
  const float* shift  = (const float*)d_in[1];
  const float* wkv_in = (const float*)d_in[2];
  const float* tmx    = (const float*)d_in[3];
  const float* tmr    = (const float*)d_in[4];
  const float* tmk    = (const float*)d_in[5];
  const float* tmv    = (const float*)d_in[6];
  const float* tmw    = (const float*)d_in[7];
  const float* tmv2   = (const float*)d_in[8];
  const float* mw1    = (const float*)d_in[9];
  const float* mw2    = (const float*)d_in[10];
  const float* tdec   = (const float*)d_in[11];
  const float* tdw1   = (const float*)d_in[12];
  const float* tdw2   = (const float*)d_in[13];
  const float* tvw1   = (const float*)d_in[14];
  const float* tvw2   = (const float*)d_in[15];
  const float* W_r    = (const float*)d_in[16];
  const float* W_k    = (const float*)d_in[17];
  const float* W_v    = (const float*)d_in[18];
  const float* W_o    = (const float*)d_in[19];
  const float* ln_w   = (const float*)d_in[20];
  const float* ln_b   = (const float*)d_in[21];

  float* out = (float*)d_out;
  float* out_shift = out + (size_t)BB * TT * DD;
  float* out_wkv   = out_shift + (size_t)BB * DD;

  const size_t MT = (size_t)BB * TT;
  char* ws = (char*)d_ws;
  size_t off = 0;
  auto alloc = [&](size_t bytes) -> char* {
    char* p = ws + off; off += (bytes + 255) & ~(size_t)255; return p;
  };
  bf16*  xr   = (bf16*)alloc(MT * DD * sizeof(bf16));
  bf16*  xk   = (bf16*)alloc(MT * DD * sizeof(bf16));
  bf16*  xv   = (bf16*)alloc(MT * DD * sizeof(bf16));
  bf16*  xv2  = (bf16*)alloc(MT * DD * sizeof(bf16));
  bf16*  wrT  = (bf16*)alloc((size_t)DD * DD * sizeof(bf16));
  bf16*  wkT  = (bf16*)alloc((size_t)DD * DD * sizeof(bf16));
  bf16*  wvT  = (bf16*)alloc((size_t)DD * DD * sizeof(bf16));
  bf16*  woT  = (bf16*)alloc((size_t)DD * DD * sizeof(bf16));
  float* dec  = (float*)alloc(MT * DD * sizeof(float));
  float* v2c  = (float*)alloc(MT * DD * sizeof(float));
  float* rb   = (float*)alloc(MT * DD * sizeof(float));
  float* kb   = (float*)alloc(MT * DD * sizeof(float));
  float* vb   = (float*)alloc(MT * DD * sizeof(float));
  float* v2b  = (float*)alloc(MT * DD * sizeof(float));
  float* yb   = (float*)alloc(MT * DD * sizeof(float));
  bf16*  lnb  = (bf16*)alloc(MT * DD * sizeof(bf16));

  dim3 tpb(16, 16);
  dim3 tgr(DD / 16, DD / 16);
  gf_transpose_bf16<<<tgr, tpb, 0, stream>>>(W_r, wrT, DD, DD);
  gf_transpose_bf16<<<tgr, tpb, 0, stream>>>(W_k, wkT, DD, DD);
  gf_transpose_bf16<<<tgr, tpb, 0, stream>>>(W_v, wvT, DD, DD);
  gf_transpose_bf16<<<tgr, tpb, 0, stream>>>(W_o, woT, DD, DD);

  gf_prep_mix<<<(int)MT, 256, 0, stream>>>(
      x, shift, tmx, tmr, tmk, tmv, tmw, tmv2, mw1, mw2,
      tdec, tdw1, tdw2, tvw1, tvw2, xr, xk, xv, xv2, dec, v2c);

  dim3 gg((int)(MT / 128), DD / 256);
  gf_gemm_bf16<<<gg, 256, 0, stream>>>(xr,  wrT, rb,  (int)MT, DD, DD, 0, nullptr);
  gf_gemm_bf16<<<gg, 256, 0, stream>>>(xk,  wkT, kb,  (int)MT, DD, DD, 1, dec);
  gf_gemm_bf16<<<gg, 256, 0, stream>>>(xv,  wvT, vb,  (int)MT, DD, DD, 0, nullptr);
  gf_gemm_bf16<<<gg, 256, 0, stream>>>(xv2, wvT, v2b, (int)MT, DD, DD, 2, v2c);

  gf_wkv_scan<<<BB * HH, 64, 0, stream>>>(rb, kb, vb, dec, v2b, wkv_in, yb, out_wkv);

  gf_layernorm<<<(int)MT, 256, 0, stream>>>(yb, ln_w, ln_b, lnb);

  gf_gemm_bf16<<<gg, 256, 0, stream>>>(lnb, woT, out, (int)MT, DD, DD, 0, nullptr);

  gf_shift_copy<<<(BB * DD + 255) / 256, 256, 0, stream>>>(x, out_shift);
}
